// MyGraphLayer_86749749445010
// MI455X (gfx1250) — matrix operand order repaired and checked by reading, not compiled, not run on hardware
//
#include <hip/hip_runtime.h>
#include <math.h>

// ---------------------------------------------------------------------------
// Sizes from the reference: B=64, N=1024, E=128, O=64, 5 forward iterations.
// ---------------------------------------------------------------------------
#define BB 64
#define NN 1024
#define EE 128
#define OO 64
#define ITERS 5

typedef unsigned short u16;
typedef unsigned long long u64;
typedef __attribute__((ext_vector_type(16))) __bf16 v16bf;
typedef __attribute__((ext_vector_type(8)))  __bf16 v8bf;
typedef __attribute__((ext_vector_type(8)))  float  v8f;

// gcc-style vector to match the async-to-LDS builtin prototype:
//   void (*)(v4i32 addrspace(1)*, v4i32 addrspace(3)*, imm, imm)
typedef int v4i32 __attribute__((vector_size(16)));
typedef __attribute__((address_space(1))) v4i32 gv4i32;
typedef __attribute__((address_space(3))) v4i32 lv4i32;

#if __has_builtin(__builtin_amdgcn_global_load_async_to_lds_b128)
#define USE_ASYNC 1
#else
#define USE_ASYNC 0
#endif

// float -> bf16 (round-to-nearest-even)
__device__ __forceinline__ u16 f2bf(float x) {
    unsigned int u = __builtin_bit_cast(unsigned int, x);
    unsigned int r = (u + 0x7FFFu + ((u >> 16) & 1u)) >> 16;
    return (u16)r;
}
__device__ __forceinline__ __bf16 bfc(float x) {
    u16 s = f2bf(x);
    return __builtin_bit_cast(__bf16, s);
}

// Async 16-byte global -> LDS copy (per-lane). Global flat == as(1) values;
// LDS generic pointers carry the LDS byte offset in addr[31:0] (ISA 10.2).
__device__ __forceinline__ void async_cp16(const u16* g, u16* l) {
#if USE_ASYNC
    __builtin_amdgcn_global_load_async_to_lds_b128(
        (gv4i32*)(unsigned long long)(size_t)g,
        (lv4i32*)(unsigned int)(size_t)l,
        0, 0);
#else
    (void)g; (void)l;
#endif
}

__device__ __forceinline__ void wait_async() {
#if __has_builtin(__builtin_amdgcn_s_wait_asynccnt)
    __builtin_amdgcn_s_wait_asynccnt(0);
#elif defined(__AMDGCN__)
    asm volatile("s_wait_asynccnt 0x0" ::: "memory");
#endif
}

// A-fragment for v_wmma_f32_16x16x32_bf16 (16x32, 16-bit):
// lane (m = lane%16, h = lane/16) holds K in [h*8, h*8+8) and [16+h*8, 16+h*8+8)
// -> vector elems 0..7 = first chunk, 8..15 = second chunk (2 bf16 per VGPR).
__device__ __forceinline__ v16bf fragA(const u16* row, int h) {
    v8bf lo = *(const v8bf*)(row + h * 8);
    v8bf hi = *(const v8bf*)(row + h * 8 + 16);
    v16bf r;
#pragma unroll
    for (int i = 0; i < 8; ++i) { r[i] = lo[i]; r[i + 8] = hi[i]; }
    return r;
}

// B-fragment (32x16, 16-bit): lane (n = lane%16, h = lane/16) holds K in
// [h*8, h*8+8) and [16+h*8, 16+h*8+8) of column n. `p` points at K=h*8.
__device__ __forceinline__ v16bf fragB(const u16* p) {
    v8bf lo = *(const v8bf*)(p);
    v8bf hi = *(const v8bf*)(p + 16);
    v16bf r;
#pragma unroll
    for (int i = 0; i < 8; ++i) { r[i] = lo[i]; r[i + 8] = hi[i]; }
    return r;
}

// ---------------------------------------------------------------------------
// CFG (f32) -> bf16, 4 elements / thread.
// ---------------------------------------------------------------------------
__global__ __launch_bounds__(256) void cvt_cfg(const float* __restrict__ src,
                                               u16* __restrict__ dst) {
    size_t i = (size_t)blockIdx.x * blockDim.x + threadIdx.x;  // element-quad
    float4 f = ((const float4*)src)[i];
    u64 packed = (u64)f2bf(f.x) | ((u64)f2bf(f.y) << 16) |
                 ((u64)f2bf(f.z) << 32) | ((u64)f2bf(f.w) << 48);
    *(u64*)(dst + i * 4) = packed;
}

// ---------------------------------------------------------------------------
// emb = relu(INPUT) -> f32 state + transposed bf16 copy embT[b][e][n]
// ---------------------------------------------------------------------------
__global__ __launch_bounds__(256) void init_emb(const float* __restrict__ input,
                                                float* __restrict__ emb,
                                                u16* __restrict__ embT) {
    size_t i = (size_t)blockIdx.x * blockDim.x + threadIdx.x;  // B*N*E total
    float x = input[i];
    x = x > 0.0f ? x : 0.0f;
    emb[i] = x;
    size_t b = i / ((size_t)NN * EE);
    size_t rem = i % ((size_t)NN * EE);
    size_t n = rem / EE;
    size_t e = rem % EE;
    embT[b * (size_t)EE * NN + e * NN + n] = f2bf(x);
}

// ---------------------------------------------------------------------------
// W1/W2 (f32, [k][n]) -> transposed bf16 [n][k]
// ---------------------------------------------------------------------------
__global__ __launch_bounds__(256) void cvt_w(const float* __restrict__ W1,
                                             const float* __restrict__ W2,
                                             u16* __restrict__ w1t,
                                             u16* __restrict__ w2t) {
    int i = blockIdx.x * blockDim.x + threadIdx.x;  // E*E total
    int k = i / EE;
    int n = i % EE;
    w1t[n * EE + k] = f2bf(W1[i]);
    w2t[n * EE + k] = f2bf(W2[i]);
}

// ---------------------------------------------------------------------------
// t = CFG_bf16 @ emb_bf16   (per batch: [1024x1024] x [1024x128] -> f32)
// Block: 256 threads (8 waves), 128x128 output tile, K-step 32.
// Double-buffered LDS; next tile staged (async-to-LDS when available, else
// global->regs->LDS split around the compute) while current tile is in WMMA.
// ---------------------------------------------------------------------------
__global__ __launch_bounds__(256) void gemm_msg(const u16* __restrict__ cfg,
                                                const u16* __restrict__ embT,
                                                float* __restrict__ t_out) {
    const int blk  = blockIdx.x;          // BB * (NN/128)
    const int b    = blk >> 3;            // NN/128 == 8
    const int rt   = blk & 7;
    const int tid  = threadIdx.x;
    const int lane = tid & 31;
    const int wave = tid >> 5;

    __shared__ u16 ldsA[2][128][40];   // 128 rows x 32 K (80B rows, 16B aligned)
    __shared__ u16 ldsBT[2][128][40];  // 128 cols x 32 K

    const u16* cfgB  = cfg  + (size_t)b * NN * NN + (size_t)(rt * 128) * NN;
    const u16* embTB = embT + (size_t)b * EE * NN;

    v8f acc[8] = {};

    const int m = lane & 15;
    const int h = lane >> 4;

    // per-thread staging coordinates: 2 threads per row, 16B chunks
    const int r = tid >> 1;             // 0..127
    const int c = (tid & 1) * 16;       // 0 or 16
    const u16* srcA = cfgB  + (size_t)r * NN + c;
    const u16* srcB = embTB + (size_t)r * NN + c;

    // ---- prologue: stage tile k0=0 into buffer 0 ----
#if USE_ASYNC
    async_cp16(srcA,     &ldsA[0][r][c]);
    async_cp16(srcA + 8, &ldsA[0][r][c + 8]);
    async_cp16(srcB,     &ldsBT[0][r][c]);
    async_cp16(srcB + 8, &ldsBT[0][r][c + 8]);
    wait_async();
#else
    *(v8bf*)&ldsA[0][r][c]      = *(const v8bf*)(srcA);
    *(v8bf*)&ldsA[0][r][c + 8]  = *(const v8bf*)(srcA + 8);
    *(v8bf*)&ldsBT[0][r][c]     = *(const v8bf*)(srcB);
    *(v8bf*)&ldsBT[0][r][c + 8] = *(const v8bf*)(srcB + 8);
#endif
    __syncthreads();

    for (int k0 = 0; k0 < NN; k0 += 32) {
        const int cur = (k0 >> 5) & 1;
        const int nxt = cur ^ 1;
        const bool pre = (k0 + 32) < NN;

#if USE_ASYNC
        if (pre) {  // async copy of next tile straight into the other buffer
            async_cp16(srcA + k0 + 32, &ldsA[nxt][r][c]);
            async_cp16(srcA + k0 + 40, &ldsA[nxt][r][c + 8]);
            async_cp16(srcB + k0 + 32, &ldsBT[nxt][r][c]);
            async_cp16(srcB + k0 + 40, &ldsBT[nxt][r][c + 8]);
        }
#else
        v8bf ra0, ra1, rb0, rb1;  // issue next-tile global loads before compute
        if (pre) {
            ra0 = *(const v8bf*)(srcA + k0 + 32);
            ra1 = *(const v8bf*)(srcA + k0 + 40);
            rb0 = *(const v8bf*)(srcB + k0 + 32);
            rb1 = *(const v8bf*)(srcB + k0 + 40);
        }
#endif

        // ---- compute on buffer `cur` ----
        const v16bf a = fragA(&ldsA[cur][wave * 16 + m][0], h);
#pragma unroll
        for (int nt = 0; nt < 8; ++nt) {
            const v16bf bf = fragB(&ldsBT[cur][nt * 16 + m][h * 8]);
            acc[nt] = __builtin_amdgcn_wmma_f32_16x16x32_bf16(
                false, a, false, bf, (short)0, acc[nt], false, false);
        }

#if USE_ASYNC
        if (pre) wait_async();
#else
        if (pre) {  // LDS stores after compute (LDS is in-order per wave)
            *(v8bf*)&ldsA[nxt][r][c]      = ra0;
            *(v8bf*)&ldsA[nxt][r][c + 8]  = ra1;
            *(v8bf*)&ldsBT[nxt][r][c]     = rb0;
            *(v8bf*)&ldsBT[nxt][r][c + 8] = rb1;
        }
#endif
        __syncthreads();
    }

    // store: C layout -> row M = v + 8*(lane/16), col = lane%16 within tile
    float* outB = t_out + (size_t)b * NN * EE + (size_t)(rt * 128) * EE;
    const int rbase = wave * 16 + (h << 3);
#pragma unroll
    for (int nt = 0; nt < 8; ++nt) {
#pragma unroll
        for (int v = 0; v < 8; ++v) {
            outB[(size_t)(rbase + v) * EE + nt * 16 + m] = acc[nt][v];
        }
    }
}

// ---------------------------------------------------------------------------
// Fused MLP + residual:
//   t1 = relu(t @ W1 + b1); t2 = relu(t1 @ W2 + b2); emb = tanh(emb + t2)
// Flattened rows [B*N, E]; block handles 128 rows, wave w rows w*16..+15.
// Weights read as bf16 B-fragments straight from global (L2-resident, 64 KB).
// ---------------------------------------------------------------------------
__global__ __launch_bounds__(256) void mlp_fuse(const float* __restrict__ t_in,
                                                const u16* __restrict__ w1t,
                                                const u16* __restrict__ w2t,
                                                const float* __restrict__ b1,
                                                const float* __restrict__ b2,
                                                float* __restrict__ emb,
                                                u16* __restrict__ embT) {
    const int blk  = blockIdx.x;          // (B*N)/128
    const int tid  = threadIdx.x;
    const int lane = tid & 31;
    const int wave = tid >> 5;
    const int m    = lane & 15;
    const int h    = lane >> 4;

    __shared__ u16 lt1[128][136];  // bf16 t1 staging (272B rows, 16B aligned)

    const float* trow = t_in + (size_t)blk * 128 * EE;
    const int lrow = wave * 16 + m;

    // A fragments of t (f32 -> bf16), 4 K-chunks of 32
    v16bf afr[4];
#pragma unroll
    for (int ks = 0; ks < 4; ++ks) {
        const float* p = trow + (size_t)lrow * EE + ks * 32;
        float4 f0a = *(const float4*)(p + h * 8);
        float4 f0b = *(const float4*)(p + h * 8 + 4);
        float4 f1a = *(const float4*)(p + 16 + h * 8);
        float4 f1b = *(const float4*)(p + 16 + h * 8 + 4);
        v16bf a;
        a[0] = bfc(f0a.x);  a[1] = bfc(f0a.y);  a[2] = bfc(f0a.z);  a[3] = bfc(f0a.w);
        a[4] = bfc(f0b.x);  a[5] = bfc(f0b.y);  a[6] = bfc(f0b.z);  a[7] = bfc(f0b.w);
        a[8] = bfc(f1a.x);  a[9] = bfc(f1a.y);  a[10] = bfc(f1a.z); a[11] = bfc(f1a.w);
        a[12] = bfc(f1b.x); a[13] = bfc(f1b.y); a[14] = bfc(f1b.z); a[15] = bfc(f1b.w);
        afr[ks] = a;
    }

    // ---- GEMM 1: t @ W1 + b1 ----
    v8f acc[8];
#pragma unroll
    for (int nt = 0; nt < 8; ++nt) {
        const float bv = b1[nt * 16 + m];
#pragma unroll
        for (int v = 0; v < 8; ++v) acc[nt][v] = bv;
    }
#pragma unroll
    for (int ks = 0; ks < 4; ++ks) {
#pragma unroll
        for (int nt = 0; nt < 8; ++nt) {
            const v16bf bf = fragB(&w1t[(nt * 16 + m) * EE + ks * 32 + h * 8]);
            acc[nt] = __builtin_amdgcn_wmma_f32_16x16x32_bf16(
                false, afr[ks], false, bf, (short)0, acc[nt], false, false);
        }
    }

    // relu -> bf16 -> LDS (row-major so it re-fragments as A for GEMM 2)
    const int rb = wave * 16 + (h << 3);
#pragma unroll
    for (int nt = 0; nt < 8; ++nt) {
#pragma unroll
        for (int v = 0; v < 8; ++v) {
            float x = acc[nt][v];
            x = x > 0.0f ? x : 0.0f;
            lt1[rb + v][nt * 16 + m] = f2bf(x);
        }
    }
    __syncthreads();

    // ---- GEMM 2: t1 @ W2 + b2 ----
    v8f acc2[8];
#pragma unroll
    for (int nt = 0; nt < 8; ++nt) {
        const float bv = b2[nt * 16 + m];
#pragma unroll
        for (int v = 0; v < 8; ++v) acc2[nt][v] = bv;
    }
#pragma unroll
    for (int ks = 0; ks < 4; ++ks) {
        const v16bf a2 = fragA(&lt1[lrow][ks * 32], h);
#pragma unroll
        for (int nt = 0; nt < 8; ++nt) {
            const v16bf bf = fragB(&w2t[(nt * 16 + m) * EE + ks * 32 + h * 8]);
            acc2[nt] = __builtin_amdgcn_wmma_f32_16x16x32_bf16(
                false, a2, false, bf, (short)0, acc2[nt], false, false);
        }
    }

    // ---- epilogue: relu, residual add, tanh; write emb (f32) and embT (bf16)
    const size_t growbase = (size_t)blk * 128 + rb;
#pragma unroll
    for (int nt = 0; nt < 8; ++nt) {
#pragma unroll
        for (int v = 0; v < 8; ++v) {
            float x = acc2[nt][v];
            x = x > 0.0f ? x : 0.0f;
            const int col = nt * 16 + m;
            const size_t g = growbase + v;          // flattened row in [B*N]
            float e = emb[g * EE + col];
            float val = tanhf(e + x);
            emb[g * EE + col] = val;
            const size_t bb = g / NN;
            const size_t n  = g % NN;
            embT[bb * (size_t)EE * NN + (size_t)col * NN + n] = f2bf(val);
        }
    }
}

// ---------------------------------------------------------------------------
// mid = sum_n emb[b,n,:];  out = mid @ wOut + bOut
// ---------------------------------------------------------------------------
__global__ __launch_bounds__(128) void pool_out(const float* __restrict__ emb,
                                                const float* __restrict__ wOut,
                                                const float* __restrict__ bOut,
                                                float* __restrict__ out) {
    const int b = blockIdx.x;
    const int tid = threadIdx.x;  // 128 threads = one per embedding channel
    __shared__ float mid[EE];
    const float* e = emb + (size_t)b * NN * EE;
    float s = 0.0f;
    for (int n = 0; n < NN; ++n) s += e[(size_t)n * EE + tid];
    mid[tid] = s;
    __syncthreads();
    if (tid < OO) {
        float o = bOut[tid];
#pragma unroll 4
        for (int k = 0; k < EE; ++k) o += mid[k] * wOut[k * OO + tid];
        out[b * OO + tid] = o;
    }
}

// ---------------------------------------------------------------------------
extern "C" void kernel_launch(void* const* d_in, const int* in_sizes, int n_in,
                              void* d_out, int out_size, void* d_ws, size_t ws_size,
                              hipStream_t stream) {
    const float* INPUT = (const float*)d_in[0];
    const float* CFG   = (const float*)d_in[1];
    const float* W1    = (const float*)d_in[2];
    const float* b1    = (const float*)d_in[3];
    const float* W2    = (const float*)d_in[4];
    const float* b2    = (const float*)d_in[5];
    const float* wOut  = (const float*)d_in[6];
    const float* bOut  = (const float*)d_in[7];
    float* out = (float*)d_out;

    // Workspace layout (all 256B-aligned by construction)
    char* p = (char*)d_ws;
    u16*   cfg_bf = (u16*)p;   p += (size_t)BB * NN * NN * sizeof(u16);   // 128 MB
    float* emb    = (float*)p; p += (size_t)BB * NN * EE * sizeof(float); //  32 MB
    u16*   embT   = (u16*)p;   p += (size_t)BB * NN * EE * sizeof(u16);   //  16 MB
    float* tbuf   = (float*)p; p += (size_t)BB * NN * EE * sizeof(float); //  32 MB
    u16*   w1t    = (u16*)p;   p += (size_t)EE * EE * sizeof(u16);
    u16*   w2t    = (u16*)p;   p += (size_t)EE * EE * sizeof(u16);
    (void)ws_size; (void)in_sizes; (void)n_in; (void)out_size;

    // one-time precision conversions (CFG reused 5x -> bf16 pays for itself)
    cvt_cfg<<<(unsigned)((size_t)BB * NN * NN / 4 / 256), 256, 0, stream>>>(CFG, cfg_bf);
    init_emb<<<(unsigned)((size_t)BB * NN * EE / 256), 256, 0, stream>>>(INPUT, emb, embT);
    cvt_w<<<(EE * EE) / 256, 256, 0, stream>>>(W1, W2, w1t, w2t);

    for (int it = 0; it < ITERS; ++it) {
        gemm_msg<<<BB * (NN / 128), 256, 0, stream>>>(cfg_bf, embT, tbuf);
        mlp_fuse<<<(BB * NN) / 128, 256, 0, stream>>>(tbuf, w1t, w2t, b1, b2, emb, embT);
    }

    pool_out<<<BB, 128, 0, stream>>>(emb, wOut, bOut, out);
}
